// DotProduct_Layer_7954279432665
// MI455X (gfx1250) — matrix operand order repaired and checked by reading, not compiled, not run on hardware
//
#include <hip/hip_runtime.h>
#include <hip/hip_bf16.h>
#include <math.h>

// Problem constants (from reference)
#define BB   2048
#define LL   200
#define DD   256
#define PAD_VAL (-4294967295.0f)   // -2^32 + 1

typedef __bf16 v16bf __attribute__((ext_vector_type(16)));
typedef float  v8f   __attribute__((ext_vector_type(8)));
typedef float  f4v   __attribute__((ext_vector_type(4)));

static __device__ __forceinline__ unsigned short f2bf(float x) {
    __bf16 h = (__bf16)x;                       // RNE convert
    return __builtin_bit_cast(unsigned short, h);
}
static __device__ __forceinline__ float bf2f(unsigned short u) {
    unsigned int v = ((unsigned int)u) << 16;
    return __builtin_bit_cast(float, v);
}

// ---------------------------------------------------------------------------
// Kernel 1: M = W_q @ W_k^T   (256x256, tiny one-time fusion; fp32 VALU)
// ---------------------------------------------------------------------------
__global__ void __launch_bounds__(256) fuse_wq_wk(const float* __restrict__ Wq,
                                                  const float* __restrict__ Wk,
                                                  float* __restrict__ M) {
    const int i = blockIdx.x;        // row of M
    const int j = threadIdx.x;       // col of M
    const float* wq = Wq + i * DD;
    const float* wk = Wk + j * DD;   // M[i][j] = sum_e Wq[i][e]*Wk[j][e]
    float s = 0.0f;
#pragma unroll 8
    for (int e = 0; e < DD; ++e) s = fmaf(wq[e], wk[e], s);
    M[i * DD + j] = s;
}

// ---------------------------------------------------------------------------
// Kernel 2: repack a 256x256 fp32 B matrix into bf16 in WMMA B-fragment order:
//   Bpk[(((tn*8 + ks)*32 + lane)*16 + j)] = bf16( B[(ks*32 + (lane>>4)*16 + j)*256
//                                                   + tn*16 + (lane&15)] )
// so each GEMM lane loads its whole fragment as one aligned 32-byte vector.
// 4096 threads, 16 elements each (one-time, 64K elements).
// ---------------------------------------------------------------------------
__global__ void __launch_bounds__(256) repack_b(const float* __restrict__ Bm,
                                                __bf16* __restrict__ Bpk) {
    const int t    = blockIdx.x * 256 + threadIdx.x;   // 0..4095 == (tn*8+ks)*32+lane
    const int lane = t & 31;
    const int ks   = (t >> 5) & 7;
    const int tn   = t >> 8;
    const int hl   = lane >> 4;
    const int r    = lane & 15;
    const int krow = ks * 32 + hl * 16;
    const int n    = tn * 16 + r;
    __bf16* dst = Bpk + (size_t)t * 16;
#pragma unroll
    for (int j = 0; j < 16; ++j)
        dst[j] = (__bf16)Bm[(size_t)(krow + j) * DD + n];
}

// ---------------------------------------------------------------------------
// Kernel 3: C[M x 256] = A[M x 256] @ B[256 x 256] via v_wmma_f32_16x16x32_bf16
// A: fp32, converted on the fly. B: pre-packed bf16 fragments (see repack_b).
// One wave per 16x16 C tile; 8 waves/block; K loop fully unrolled (8 WMMAs).
// ---------------------------------------------------------------------------
__global__ void __launch_bounds__(256) gemm_wmma_bf16(const float* __restrict__ A,
                                                      const __bf16* __restrict__ Bpk,
                                                      float* __restrict__ C,
                                                      int Mrows) {
    const int N = DD, K = DD;
    const int wave = threadIdx.x >> 5;
    const int lane = threadIdx.x & 31;
    const int hl   = lane >> 4;          // lane half (0/1)
    const int r    = lane & 15;

    const int tiles_n = N / 16;          // 16
    const int tile = blockIdx.x * 8 + wave;
    const int tm = tile / tiles_n;
    const int tn = tile % tiles_n;
    if (tm * 16 >= Mrows) return;        // wave-uniform

    const int m = tm * 16 + r;           // A row for this lane

    v8f acc = {};
#pragma unroll
    for (int k0 = 0; k0 < K; k0 += 32) {
        v16bf a;
        const float* arow = A + (size_t)m * K + k0;
        // A fragment in this code's k order (B packed to match): elems 0..15 -> K = hl*16+j
        const f4v a0 = *(const f4v*)(arow + hl * 16);
        const f4v a1 = *(const f4v*)(arow + hl * 16 + 4);
        const f4v a2 = *(const f4v*)(arow + hl * 16 + 8);
        const f4v a3 = *(const f4v*)(arow + hl * 16 + 12);
#pragma unroll
        for (int j = 0; j < 4; ++j) {
            a[j]      = (__bf16)a0[j];
            a[4 + j]  = (__bf16)a1[j];
            a[8 + j]  = (__bf16)a2[j];
            a[12 + j] = (__bf16)a3[j];
        }
        // B fragment: one aligned 32-byte vector load per lane
        const v16bf b = *(const v16bf*)(Bpk +
            ((size_t)((tn * 8 + (k0 >> 5)) * 32 + lane)) * 16);

        acc = __builtin_amdgcn_wmma_f32_16x16x32_bf16(
            /*neg_a=*/false, a, /*neg_b=*/false, b,
            /*c_mod=*/(short)0, acc, /*reuse_a=*/false, /*reuse_b=*/false);
    }
    // C/D f32 16x16: element (hl*8 + rr, r) in vgpr rr
#pragma unroll
    for (int rr = 0; rr < 8; ++rr) {
        C[(size_t)(tm * 16 + hl * 8 + rr) * N + tn * 16 + r] = acc[rr];
    }
}

// ---------------------------------------------------------------------------
// Kernel 4: fused per-batch attention (the HBM-roofline part).
// One block per batch. k[b] (200x256 f32, 200KB) is streamed from HBM ONCE
// with non-temporal loads, staged in LDS as bf16 with padded rows (130 dwords)
// -> bank-conflict-free for both row sweeps (scores) and column sweeps (out).
//   score[l] = u[b].k[b,l] + bias*256 ; sigmoid ; mask->PAD ; /16 ; softmax
//   Wout[b]  = sum_l a[l] * k[b,l]
// ---------------------------------------------------------------------------
__global__ void __launch_bounds__(256) attn_fused(const float* __restrict__ Kin,
                                                  const float* __restrict__ U,
                                                  const int*   __restrict__ keys_len,
                                                  const float* __restrict__ bias,
                                                  float* __restrict__ Wout) {
    const int b   = blockIdx.x;
    const int tid = threadIdx.x;
    const int RSDW = 130;   // padded row stride in dwords (260 bf16)

    extern __shared__ char smem_raw[];
    unsigned int* kdw = (unsigned int*)smem_raw;                 // 200*130 dwords = 104000 B
    float* u_sh = (float*)(smem_raw + (size_t)LL * RSDW * 4);    // 256 f
    float* red  = u_sh + DD;                                     // 256 f
    float* a_w  = red + 256;                                     // 256 f

    // u vector for this batch
    u_sh[tid] = U[(size_t)b * DD + tid];

    // Stream k[b] from HBM (non-temporal float4), convert to bf16, store b64
    const float* kb = Kin + (size_t)b * LL * DD;
    for (int i = tid; i < LL * (DD / 4); i += 256) {
        const int row = i >> 6;          // / (DD/4)
        const int c4  = i & 63;
        const f4v f = __builtin_nontemporal_load((const f4v*)(kb + row * DD + c4 * 4));
        const unsigned int p0 = ((unsigned int)f2bf(f[1]) << 16) | f2bf(f[0]);
        const unsigned int p1 = ((unsigned int)f2bf(f[3]) << 16) | f2bf(f[2]);
        const unsigned int idx = row * RSDW + c4 * 2;
        kdw[idx]     = p0;
        kdw[idx + 1] = p1;
    }
    __syncthreads();

    // ---- scores: thread t owns row t (t < 200) ----
    float x;
    if (tid < LL) {
        float s = 0.0f;
        const unsigned int* krow = kdw + tid * RSDW;
#pragma unroll 4
        for (int j = 0; j < DD / 2; ++j) {
            const unsigned int p = krow[j];
            s = fmaf(bf2f((unsigned short)(p & 0xffffu)), u_sh[2 * j],     s);
            s = fmaf(bf2f((unsigned short)(p >> 16)),     u_sh[2 * j + 1], s);
        }
        s += bias[0] * (float)DD;
        const float sig = 1.0f / (1.0f + __expf(-s));
        x = (tid < keys_len[b]) ? sig : PAD_VAL;   // mask BEFORE the /16, like ref
        x *= 0.0625f;                              // / sqrt(256)
    } else {
        x = -INFINITY;                             // excluded from softmax
    }

    // softmax over the 200 valid lanes (block tree-reduce: max, then sum)
    red[tid] = x;
    __syncthreads();
#pragma unroll
    for (int st = 128; st > 0; st >>= 1) {
        if (tid < st) red[tid] = fmaxf(red[tid], red[tid + st]);
        __syncthreads();
    }
    const float mx = red[0];
    __syncthreads();
    const float e = __expf(x - mx);                // exp(-inf - finite) == 0
    a_w[tid] = e;
    red[tid] = e;
    __syncthreads();
#pragma unroll
    for (int st = 128; st > 0; st >>= 1) {
        if (tid < st) red[tid] += red[tid + st];
        __syncthreads();
    }
    const float inv = 1.0f / red[0];               // all-masked => uniform 1/200, like ref

    // ---- weighted sum: thread owns column d = tid; a_w reads broadcast ----
    const int dw = tid >> 1;
    const int hi = tid & 1;
    float w = 0.0f;
    for (int l = 0; l < LL; ++l) {
        const unsigned int p = kdw[l * RSDW + dw];
        const unsigned short h = hi ? (unsigned short)(p >> 16)
                                    : (unsigned short)(p & 0xffffu);
        w = fmaf(a_w[l], bf2f(h), w);
    }
    Wout[(size_t)b * DD + tid] = w * inv;
}

// ---------------------------------------------------------------------------
extern "C" void kernel_launch(void* const* d_in, const int* in_sizes, int n_in,
                              void* d_out, int out_size, void* d_ws, size_t ws_size,
                              hipStream_t stream) {
    const float* q    = (const float*)d_in[0];   // [2048,1,256]
    const float* k    = (const float*)d_in[1];   // [2048,200,256]
    // d_in[2] = v : unused by the reference math (vp derives from k)
    const int*   klen = (const int*)  d_in[3];   // [2048,1]
    const float* Wq   = (const float*)d_in[4];   // [256,256]
    const float* Wk   = (const float*)d_in[5];   // [256,256]
    const float* Wv   = (const float*)d_in[6];   // [256,256]
    const float* bias = (const float*)d_in[7];   // [1]
    float* out = (float*)d_out;                  // [2048,1,256]

    char* ws = (char*)d_ws;
    float*  Mfused = (float*)ws;                                    // 256KB
    float*  U      = (float*)(ws + (size_t)DD * DD * 4);            // 2MB
    float*  Wmat   = (float*)(ws + (size_t)DD * DD * 4
                                 + (size_t)BB * DD * 4);            // 2MB
    __bf16* Bpk    = (__bf16*)(ws + (size_t)DD * DD * 4
                                  + 2 * (size_t)BB * DD * 4);       // 128KB (reused)

    // 1) M = Wq @ Wk^T  (tiny)
    fuse_wq_wk<<<DD, 256, 0, stream>>>(Wq, Wk, Mfused);

    // 2) U = Q @ M  (WMMA bf16; B pre-packed into fragment layout)
    repack_b<<<16, 256, 0, stream>>>(Mfused, Bpk);
    gemm_wmma_bf16<<<(BB / 16) * (DD / 16) / 8, 256, 0, stream>>>(q, Bpk, U, BB);

    // 3) fused streaming attention: one block per batch, single HBM pass over k
    const size_t shmem = (size_t)LL * 130 * 4 + 3 * 256 * 4;        // 107,072 B
    attn_fused<<<BB, 256, shmem, stream>>>(k, U, klen, bias, Wmat);

    // 4) out = Wmat @ Wv  (WMMA bf16; same packed-B path, buffer reused)
    repack_b<<<16, 256, 0, stream>>>(Wv, Bpk);
    gemm_wmma_bf16<<<(BB / 16) * (DD / 16) / 8, 256, 0, stream>>>(Wmat, Bpk, out, BB);
}
